// TransducerBeamSearcher_88201448391525
// MI455X (gfx1250) — compile-verified, matrix operand order checked by reading, not run on hardware
//
#include <hip/hip_runtime.h>
#include <hip/hip_bf16.h>
#include <math.h>

// ---------------- CDNA5 WMMA types ----------------
typedef __attribute__((ext_vector_type(16))) _Float16 v16h;
typedef __attribute__((ext_vector_type(8)))  float    v8f;

// ---------------- problem dims (fixed by reference) ----------------
constexpr int kH  = 1024;   // hidden
constexpr int kV  = 4096;   // vocab
constexpr int kB  = 8;      // batch
constexpr int kT  = 128;    // frames
constexpr int kK  = 5;      // beam
constexpr int kBK = 40;     // B*K rows per step
constexpr int kMP = 48;     // M padded to 3 WMMA tiles
constexpr int kKC = 2048;   // combined K for [emb|h] @ [W_ih^T ; W_hh^T]
constexpr int kNT = 2;      // n-tiles per wave (A reuse + 2 independent WMMA chains)
constexpr float kNEG = -1.0e9f;

__device__ __forceinline__ float sigf(float x) { return 1.0f / (1.0f + __expf(-x)); }

// ---------------- WMMA fragment index helpers ----------------
// A fragment (16x32 f16), ISA layout: lane l = row(m&15) + 16*g, where
//  g=0 lanes hold k_local {0..7,16..23}, g=1 lanes hold {8..15,24..31}:
//  v<4 : k_local = g*8 + 2v + h ;  v>=4 : k_local = 16 + g*8 + 2(v-4) + h
// Packed storage: frag[((m_tile*KT + k_tile)*32 + lane)*16 + 2v+h]
__device__ __forceinline__ size_t apack_idx(int m, int k, int KT) {
    const int mt = m >> 4, r = m & 15;
    const int kt = k >> 5, kl = k & 31;
    const int hi = kl >> 4;          // upper 16 of k_local
    const int kk = kl & 15;
    const int g  = kk >> 3;          // lane group
    const int wi = kk & 7;
    const int v  = hi * 4 + (wi >> 1);
    const int h  = wi & 1;
    const int lane = r + g * 16;
    return ((size_t)(mt * KT + kt) * 32 + lane) * 16 + 2 * v + h;
}

// ---------------- B-matrix pre-pack into WMMA lane layout ----------------
// Packed layout: dst[((n_tile*(K/32)+k_tile)*32 + lane)*16 + idx]
// transposed==1: B[k][n] = src[n*ld + k]  (for W_ih^T / W_hh^T)
// transposed==0: B[k][n] = src[k*ld + n]
__global__ void pack_b(const float* __restrict__ src0, const float* __restrict__ src1,
                       int ksplit, int K, int N, int transposed, int ld,
                       _Float16* __restrict__ dst) {
    const int id = blockIdx.x * blockDim.x + threadIdx.x;
    const int total = (N >> 4) * (K >> 5) * 32;
    if (id >= total) return;
    const int lane = id & 31;
    const int frag = id >> 5;
    const int KT = K >> 5;
    const int kt = frag % KT;
    const int nt = frag / KT;
    const int n = nt * 16 + (lane & 15);
    const int g = ((lane >> 4) & 1) << 3;
    _Float16* d = dst + (size_t)id * 16;
#pragma unroll
    for (int v = 0; v < 8; ++v) {
        const int kk = (v < 4) ? (v * 2) : (16 + (v - 4) * 2);
#pragma unroll
        for (int hh = 0; hh < 2; ++hh) {
            int k = kt * 32 + g + kk + hh;
            const float* s = src0;
            if (k >= ksplit) { s = src1; k -= ksplit; }
            const float val = transposed ? s[(size_t)n * ld + k]
                                         : s[(size_t)k * ld + n];
            d[2 * v + hh] = (_Float16)val;
        }
    }
}

// One-time A pre-pack: row-major f32 [M x K] -> packed fp16 A fragments
__global__ void pack_a(const float* __restrict__ src, int M, int K,
                       _Float16* __restrict__ dst) {
    const int i = blockIdx.x * blockDim.x + threadIdx.x;
    if (i >= M * K) return;
    const int m = i / K, k = i % K;
    dst[apack_idx(m, k, K >> 5)] = (_Float16)src[(size_t)m * K + k];
}

// ---------------- generic WMMA GEMM (kNT 16x16 tiles per wave32) ----------------
// Both operands pre-packed fp16 fragments. Inner loop: 1 A-vec load, kNT B-vec
// loads, kNT independent v_wmma chains.
// mode 0: outf = acc                      (tn_proj)
// mode 1: outf = acc + bias[n]            (gates, logits)
// mode 2: outh[apack(m,n)] = tanh(acc + tn_t[b*T*H+n] + bias[n]), m<40  (joint)
__global__ void wmma_gemm(const _Float16* __restrict__ Ap,
                          const _Float16* __restrict__ Bp, int KT,
                          float* __restrict__ outf, _Float16* __restrict__ outh,
                          int ldo, int mode, const float* __restrict__ bias,
                          const float* __restrict__ tn_t, int packKT) {
    const int lane   = threadIdx.x & 31;
    const int wave   = threadIdx.x >> 5;
    const int nbase  = (blockIdx.x * (blockDim.x >> 5) + wave) * kNT;
    const int m_tile = blockIdx.y;

    v8f acc[kNT];
#pragma unroll
    for (int tq = 0; tq < kNT; ++tq) acc[tq] = {};

    const _Float16* abase = Ap + ((size_t)m_tile * KT * 32 + lane) * 16;
    const _Float16* bbase[kNT];
#pragma unroll
    for (int tq = 0; tq < kNT; ++tq)
        bbase[tq] = Bp + ((size_t)(nbase + tq) * KT * 32 + lane) * 16;

    for (int kt = 0; kt < KT; ++kt) {
        const size_t fo = (size_t)kt * 32 * 16;   // 512 halves per fragment step
        __builtin_prefetch(bbase[0] + fo + 512, 0, 1);  // global_prefetch_b8
        v16h a = *(const v16h*)(abase + fo);
#pragma unroll
        for (int tq = 0; tq < kNT; ++tq) {
            v16h b = *(const v16h*)(bbase[tq] + fo);
            acc[tq] = __builtin_amdgcn_wmma_f32_16x16x32_f16(
                /*neg_a=*/false, a, /*neg_b=*/false, b,
                /*c_mod=*/(short)0, acc[tq], /*reuse_a=*/false, /*reuse_b=*/false);
        }
    }

    // C layout: VGPR r, lane l -> m = r + (l>>4)*8, n = l&15
    const int m0 = m_tile * 16 + (((lane >> 4) & 1) << 3);
#pragma unroll
    for (int tq = 0; tq < kNT; ++tq) {
        const int n = (nbase + tq) * 16 + (lane & 15);
#pragma unroll
        for (int r = 0; r < 8; ++r) {
            const int m = m0 + r;
            const float v = acc[tq][r];
            if (mode == 0) {
                outf[(size_t)m * ldo + n] = v;
            } else if (mode == 1) {
                outf[(size_t)m * ldo + n] = v + bias[n];
            } else {
                if (m < kBK) {
                    const int bb = m / kK;
                    outh[apack_idx(m, n, packKT)] = (_Float16)tanhf(
                        v + tn_t[(size_t)bb * (kT * kH) + n] + bias[n]);
                }
            }
        }
    }
}

// ---------------- per-step elementwise kernels ----------------
// Builds packed-fp16 A = [emb | h] for the gates GEMM (K = 2048, KT = 64).
__global__ void build_A(const float* __restrict__ E, const float* __restrict__ h_cur,
                        const int* __restrict__ tokens, _Float16* __restrict__ Ap) {
    const int i = blockIdx.x * blockDim.x + threadIdx.x;
    if (i >= kBK * kKC) return;
    const int m = i >> 11;           // / 2048
    const int k = i & (kKC - 1);
    float v;
    if (k < kH) v = E[(size_t)tokens[m] * kH + k];
    else        v = h_cur[(size_t)m * kH + (k - kH)];
    Ap[apack_idx(m, k, kKC >> 5)] = (_Float16)v;
}

// LSTM cell; writes f32 state (for beam gather) + packed-fp16 h for joint GEMM.
__global__ void lstm_cell(const float* __restrict__ gates, const float* __restrict__ c_in,
                          float* __restrict__ h_cell, float* __restrict__ c_cell,
                          _Float16* __restrict__ h_pack) {
    const int idx = blockIdx.x * blockDim.x + threadIdx.x;
    if (idx >= kBK * kH) return;
    const int m = idx >> 10, n = idx & (kH - 1);
    const float* g = gates + (size_t)m * (4 * kH);
    const float ig = sigf(g[n]);
    const float fg = sigf(g[kH + n]);
    const float gg = tanhf(g[2 * kH + n]);
    const float og = sigf(g[3 * kH + n]);
    const float c  = fg * c_in[idx] + ig * gg;
    const float h  = og * tanhf(c);
    c_cell[idx] = c;
    h_cell[(size_t)m * kH + n] = h;
    h_pack[apack_idx(m, n, kH >> 5)] = (_Float16)h;
}

// ---------------- top-K + beam update (one block per batch element) ----------------
__device__ __forceinline__ void insert5(float* s, int* id, float sc, int i) {
    if (sc < s[4] || (sc == s[4] && i > id[4])) return;
    int p = 4;
    while (p > 0 && (sc > s[p - 1] || (sc == s[p - 1] && i < id[p - 1]))) {
        s[p] = s[p - 1]; id[p] = id[p - 1]; --p;
    }
    s[p] = sc; id[p] = i;
}

__global__ void topk_update(const float* __restrict__ logits,
                            const float* __restrict__ scores_in,
                            const int* __restrict__ tokens_in,
                            const int* __restrict__ lens_in,
                            const int* __restrict__ preds_in,
                            const float* __restrict__ h_in, const float* __restrict__ c_in,
                            const float* __restrict__ h_cell, const float* __restrict__ c_cell,
                            float* __restrict__ scores_out, int* __restrict__ tokens_out,
                            int* __restrict__ lens_out, int* __restrict__ preds_out,
                            float* __restrict__ h_out, float* __restrict__ c_out) {
    const int b = blockIdx.x;
    const int t = threadIdx.x;               // 256 threads
    __shared__ float red[256];
    __shared__ float s_lse[kK];
    __shared__ float ts[256 * kK];
    __shared__ int   ti[256 * kK];
    __shared__ int   s_parent[kK], s_tok[kK], s_blank[kK], s_lensp[kK];

    // per-beam log-sum-exp over V
    for (int k = 0; k < kK; ++k) {
        const float* row = logits + (size_t)(b * kK + k) * kV;
        float mx = -3.4e38f;
        for (int i = t; i < kV; i += 256) mx = fmaxf(mx, row[i]);
        red[t] = mx; __syncthreads();
        for (int s = 128; s; s >>= 1) { if (t < s) red[t] = fmaxf(red[t], red[t + s]); __syncthreads(); }
        const float m = red[0]; __syncthreads();
        float se = 0.0f;
        for (int i = t; i < kV; i += 256) se += __expf(row[i] - m);
        red[t] = se; __syncthreads();
        for (int s = 128; s; s >>= 1) { if (t < s) red[t] += red[t + s]; __syncthreads(); }
        if (t == 0) s_lse[k] = m + __logf(red[0]);
        __syncthreads();
    }

    // local top-5 over K*V candidates
    float ls[kK]; int li[kK];
#pragma unroll
    for (int j = 0; j < kK; ++j) { ls[j] = -3.4e38f; li[j] = 0x7fffffff; }
    for (int i = t; i < kK * kV; i += 256) {
        const int k = i >> 12;               // / 4096
        const int v = i & (kV - 1);
        const float sc = scores_in[b * kK + k] +
                         logits[(size_t)(b * kK + k) * kV + v] - s_lse[k];
        insert5(ls, li, sc, i);
    }
#pragma unroll
    for (int j = 0; j < kK; ++j) { ts[t * kK + j] = ls[j]; ti[t * kK + j] = li[j]; }
    __syncthreads();

    // tree merge of sorted 5-lists
    for (int s = 128; s; s >>= 1) {
        if (t < s) {
            float* sa = &ts[t * kK];       int* ia = &ti[t * kK];
            float* sb = &ts[(t + s) * kK]; int* ib = &ti[(t + s) * kK];
            float rs[kK]; int ri[kK];
            int i = 0, j = 0;
#pragma unroll
            for (int q = 0; q < kK; ++q) {
                bool takeA;
                if (sa[i] > sb[j]) takeA = true;
                else if (sa[i] < sb[j]) takeA = false;
                else takeA = (ia[i] <= ib[j]);
                if (takeA) { rs[q] = sa[i]; ri[q] = ia[i]; ++i; }
                else       { rs[q] = sb[j]; ri[q] = ib[j]; ++j; }
            }
#pragma unroll
            for (int q = 0; q < kK; ++q) { sa[q] = rs[q]; ia[q] = ri[q]; }
        }
        __syncthreads();
    }

    if (t == 0) {
#pragma unroll
        for (int j = 0; j < kK; ++j) {
            const int idx = ti[j];
            s_parent[j] = idx >> 12;
            s_tok[j]    = idx & (kV - 1);
            s_blank[j]  = (s_tok[j] == 0);
            scores_out[b * kK + j] = ts[j];
        }
    }
    __syncthreads();

    if (t < kK) {
        const int p  = s_parent[t];
        const int lp = lens_in[b * kK + p];
        s_lensp[t]   = lp;
        tokens_out[b * kK + t] = s_blank[t] ? tokens_in[b * kK + p] : s_tok[t];
        lens_out[b * kK + t]   = lp + (s_blank[t] ? 0 : 1);
    }
    __syncthreads();

    // preds gather + emit
    for (int i = t; i < kK * kT; i += 256) {
        const int k = i >> 7, pos = i & (kT - 1);
        const int p = s_parent[k];
        int val = preds_in[(b * kK + p) * kT + pos];
        if (!s_blank[k] && pos == s_lensp[k]) val = s_tok[k];
        preds_out[(b * kK + k) * kT + pos] = val;
    }
    // decoder state gather (blank keeps pre-step state)
    for (int i = t; i < kK * kH; i += 256) {
        const int k = i >> 10, n = i & (kH - 1);
        const int p = s_parent[k];
        const size_t src = (size_t)(b * kK + p) * kH + n;
        const size_t dst = (size_t)(b * kK + k) * kH + n;
        h_out[dst] = s_blank[k] ? h_in[src] : h_cell[src];
        c_out[dst] = s_blank[k] ? c_in[src] : c_cell[src];
    }
}

// ---------------- init / finalize ----------------
__global__ void zero_f32(float* p, size_t n) {
    size_t i = (size_t)blockIdx.x * blockDim.x + threadIdx.x;
    const size_t st = (size_t)gridDim.x * blockDim.x;
    for (; i < n; i += st) p[i] = 0.0f;
}

__global__ void init_scores(float* scores) {
    const int i = threadIdx.x;
    if (i < 2 * kBK) scores[i] = ((i % kK) == 0) ? 0.0f : kNEG;
}

__global__ void finalize(const float* __restrict__ scores, const int* __restrict__ lens,
                         const int* __restrict__ preds, float* __restrict__ out) {
    const int t = threadIdx.x;
    __shared__ float norm[kBK];
    __shared__ int best[kB];
    __shared__ float mx[kB];
    if (t < kBK) norm[t] = scores[t] / (float)(lens[t] + 1);
    __syncthreads();
    if (t < kB) {
        float m = norm[t * kK]; int bi = 0;
        for (int k = 1; k < kK; ++k)
            if (norm[t * kK + k] > m) { m = norm[t * kK + k]; bi = k; }
        best[t] = bi; mx[t] = m;
    }
    __syncthreads();
    // layout: [0,1024) best_preds ; [1024,1032) best_lens ; [1032] mean_score ; [1033,1073) norm
    for (int i = t; i < kB * kT; i += blockDim.x) {
        const int b = i >> 7, pos = i & (kT - 1);
        out[i] = (float)preds[(b * kK + best[b]) * kT + pos];
    }
    if (t < kB) out[kB * kT + t] = (float)lens[t * kK + best[t]];
    if (t == 0) {
        float s = 0.0f;
        for (int b = 0; b < kB; ++b) s += __expf(mx[b]);
        out[kB * kT + kB] = s / (float)kB;
    }
    for (int i = t; i < kBK; i += blockDim.x) out[kB * kT + kB + 1 + i] = norm[i];
}

// ---------------- host driver ----------------
extern "C" void kernel_launch(void* const* d_in, const int* in_sizes, int n_in,
                              void* d_out, int out_size, void* d_ws, size_t ws_size,
                              hipStream_t stream) {
    const float* tn_output = (const float*)d_in[0];
    const float* E         = (const float*)d_in[1];
    const float* W_ih      = (const float*)d_in[2];
    const float* W_hh      = (const float*)d_in[3];
    const float* b_lstm    = (const float*)d_in[4];
    const float* W_tn      = (const float*)d_in[5];
    const float* W_pn      = (const float*)d_in[6];
    const float* b_joint   = (const float*)d_in[7];
    const float* W_out     = (const float*)d_in[8];
    const float* b_out     = (const float*)d_in[9];

    char* ws = (char*)d_ws;
    size_t off = 0;
    auto alloc = [&](size_t bytes) {
        size_t r = off; off = (off + bytes + 255) & ~(size_t)255; return r;
    };
    // fp16 WMMA-packed weights: ~30 MB total -> resident in 192 MB L2 across all 128 steps
    _Float16* Wc   = (_Float16*)(ws + alloc((size_t)kKC * (4 * kH) * 2)); // [emb|h] combined
    _Float16* Wpn  = (_Float16*)(ws + alloc((size_t)kH * kH * 2));
    _Float16* Wout = (_Float16*)(ws + alloc((size_t)kH * kV * 2));
    _Float16* Wtn  = (_Float16*)(ws + alloc((size_t)kH * kH * 2));
    _Float16* Tnp16= (_Float16*)(ws + alloc((size_t)kB * kT * kH * 2));   // packed tn_output A
    float* tnp     = (float*)(ws + alloc((size_t)kB * kT * kH * 4));      // f32 tn_proj
    const size_t state_begin = off;
    _Float16* Astep = (_Float16*)(ws + alloc((size_t)kMP * kKC * 2));     // packed A [emb|h]
    _Float16* h_pack= (_Float16*)(ws + alloc((size_t)kMP * kH * 2));      // packed A h_new
    _Float16* joint_p=(_Float16*)(ws + alloc((size_t)kMP * kH * 2));      // packed A joint
    float* gates  = (float*)(ws + alloc((size_t)kMP * 4 * kH * 4));
    float* h_cell = (float*)(ws + alloc((size_t)kBK * kH * 4));
    float* c_cell = (float*)(ws + alloc((size_t)kBK * kH * 4));
    float* logits = (float*)(ws + alloc((size_t)kMP * kV * 4));
    float* h_cur  = (float*)(ws + alloc((size_t)2 * kBK * kH * 4));
    float* c_cur  = (float*)(ws + alloc((size_t)2 * kBK * kH * 4));
    float* scoresb = (float*)(ws + alloc(2 * kBK * 4));
    int* tokensb   = (int*)(ws + alloc(2 * kBK * 4));
    int* lensb     = (int*)(ws + alloc(2 * kBK * 4));
    int* predsb    = (int*)(ws + alloc((size_t)2 * kBK * kT * 4));
    const size_t state_end = off;

    // zero all mutable state (padded rows 40..47 of A buffers must stay 0)
    zero_f32<<<512, 256, 0, stream>>>((float*)(ws + state_begin),
                                      (state_end - state_begin) / 4);
    init_scores<<<1, 128, 0, stream>>>(scoresb);

    // pack weights into WMMA B-fragment lane layout (once)
    {
        const int totC = ((4 * kH) >> 4) * (kKC >> 5) * 32;
        pack_b<<<(totC + 255) / 256, 256, 0, stream>>>(W_ih, W_hh, kH, kKC, 4 * kH, 1, kH, Wc);
        const int totP = (kH >> 4) * (kH >> 5) * 32;
        pack_b<<<(totP + 255) / 256, 256, 0, stream>>>(W_pn, W_pn, kH, kH, kH, 0, kH, Wpn);
        const int totO = (kV >> 4) * (kH >> 5) * 32;
        pack_b<<<(totO + 255) / 256, 256, 0, stream>>>(W_out, W_out, kH, kH, kV, 0, kV, Wout);
        pack_b<<<(totP + 255) / 256, 256, 0, stream>>>(W_tn, W_tn, kH, kH, kH, 0, kH, Wtn);
        // pack tn_output as WMMA A fragments (M = B*T = 1024, K = 1024)
        pack_a<<<((kB * kT * kH) + 255) / 256, 256, 0, stream>>>(
            tn_output, kB * kT, kH, Tnp16);
    }

    // tn_proj = tn_output @ W_tn   (M = B*T = 1024, N = 1024, K = 1024)
    wmma_gemm<<<dim3((kH / 16) / (4 * kNT), (kB * kT) / 16), 128, 0, stream>>>(
        Tnp16, Wtn, kH >> 5, tnp, nullptr, kH, 0, nullptr, nullptr, 0);

    // sequential beam-search scan: 6 kernels per frame, ping-pong beam state
    for (int t = 0; t < kT; ++t) {
        const int cur = t & 1, nxt = cur ^ 1;
        build_A<<<(kBK * kKC + 255) / 256, 256, 0, stream>>>(
            E, h_cur + (size_t)cur * kBK * kH, tokensb + cur * kBK, Astep);
        // gates = [emb|h] @ Wc + b_lstm   (M=48, N=4096, K=2048)
        wmma_gemm<<<dim3(((4 * kH) / 16) / (4 * kNT), kMP / 16), 128, 0, stream>>>(
            Astep, Wc, kKC >> 5, gates, nullptr, 4 * kH, 1, b_lstm, nullptr, 0);
        lstm_cell<<<(kBK * kH + 255) / 256, 256, 0, stream>>>(
            gates, c_cur + (size_t)cur * kBK * kH, h_cell, c_cell, h_pack);
        // joint = tanh(tn_t + h_new @ W_pn + b_joint) -> packed fp16 (M=48,N=1024,K=1024)
        wmma_gemm<<<dim3((kH / 16) / (4 * kNT), kMP / 16), 128, 0, stream>>>(
            h_pack, Wpn, kH >> 5, nullptr, joint_p, kH, 2, b_joint,
            tnp + (size_t)t * kH, kH >> 5);
        // logits = joint @ W_out + b_out   (M=48, N=4096, K=1024)
        wmma_gemm<<<dim3((kV / 16) / (4 * kNT), kMP / 16), 128, 0, stream>>>(
            joint_p, Wout, kH >> 5, logits, nullptr, kV, 1, b_out, nullptr, 0);
        topk_update<<<kB, 256, 0, stream>>>(
            logits,
            scoresb + cur * kBK, tokensb + cur * kBK, lensb + cur * kBK,
            predsb + cur * kBK * kT,
            h_cur + (size_t)cur * kBK * kH, c_cur + (size_t)cur * kBK * kH,
            h_cell, c_cell,
            scoresb + nxt * kBK, tokensb + nxt * kBK, lensb + nxt * kBK,
            predsb + nxt * kBK * kT,
            h_cur + (size_t)nxt * kBK * kH, c_cur + (size_t)nxt * kBK * kH);
    }

    const int fin = kT & 1;  // final state parity (kT=128 -> buffer 0)
    finalize<<<1, 128, 0, stream>>>(scoresb + fin * kBK, lensb + fin * kBK,
                                    predsb + fin * kBK * kT, (float*)d_out);
    (void)in_sizes; (void)n_in; (void)out_size; (void)ws_size;
}
